// ConvGRU_85194971283736
// MI455X (gfx1250) — matrix operand order repaired and checked by reading, not compiled, not run on hardware
//
#include <hip/hip_runtime.h>
#include <string.h>

// ---------------------------------------------------------------------------
// CDNA5 (gfx1250) ConvGRU: bf16 WMMA sparse convs, async global->LDS gathers,
// fused GRU epilogues.
// ---------------------------------------------------------------------------

typedef __attribute__((ext_vector_type(16))) __bf16 v16bf;
typedef __attribute__((ext_vector_type(8)))  float  v8f;

#define ROWS_TILE 64          // voxels per workgroup
#define CIN       448
#define COUT      128
#define LDS_STRIDE 456        // 448 + 8 pad (bf16 units) -> conflict-friendly

union Frag16 {
  v16bf v;
  unsigned int u[8];
  uint2 u2[4];
  uint4 u4[2];
};

// A fragment (16x32 bf16) from LDS row-major tile, per ISA 16-bit A layout:
// lane m=l%16 is row; VGPR j holds K pair at (j&3)*2 + (j>>2)*16 + (l>>4)*8.
__device__ inline v16bf load_afrag(const __bf16* As, int rowBase, int kk, int lane) {
  int m = lane & 15, h = lane >> 4;
  const __bf16* p = As + (size_t)(rowBase + m) * LDS_STRIDE + kk * 32 + h * 8;
  Frag16 f;
  f.u2[0] = *(const uint2*)(p);        // K +0..3
  f.u2[1] = *(const uint2*)(p + 4);    // K +4..7
  f.u2[2] = *(const uint2*)(p + 16);   // K +16..19
  f.u2[3] = *(const uint2*)(p + 20);   // K +20..23
  return f.v;
}

// B fragment (32x16 bf16) from pre-packed weights: 32 B contiguous per lane.
__device__ inline v16bf load_bfrag(const unsigned int* __restrict__ wpk,
                                   int k, int kk, int ct, int lane) {
  const uint4* p = (const uint4*)(wpk +
      ((((size_t)k * 14 + kk) * 8 + ct) * 32 + (size_t)lane) * 8);
  Frag16 f;
  f.u4[0] = p[0];
  f.u4[1] = p[1];
  return f.v;
}

// Gather one 64-row tile for kernel-offset k into LDS using async copies.
// Each thread owns a quarter-row (112 bf16 = 14 x 16B). Valid rows go through
// GLOBAL_LOAD_ASYNC_TO_LDS_B128 (ASYNCcnt, no VGPR staging); masked/OOB rows
// are zeroed with plain ds stores. Caller must s_wait_asynccnt + barrier.
__device__ inline void gather_tile_async(const __bf16* __restrict__ feat,
                                         const int* __restrict__ nbr_idx,
                                         const unsigned char* __restrict__ nbr_mask,
                                         __bf16* As, int k, int tile, int tid,
                                         int Ntot) {
  int row = tid >> 2, part = tid & 3;
  int grow = tile + row;
  __bf16* dstp = As + (size_t)row * LDS_STRIDE + part * 112;
  bool valid = (grow < Ntot) && (nbr_mask[(size_t)k * Ntot + grow] != 0);
  if (valid) {
    int gi = nbr_idx[(size_t)k * Ntot + grow];
    // gfx1250 flat LDS pointer: low 32 bits are the wave-relative LDS offset.
    unsigned int ldsoff = (unsigned int)(size_t)dstp;
    unsigned long long ga =
        (unsigned long long)(size_t)(feat + (size_t)gi * CIN + part * 112);
#pragma unroll
    for (int i = 0; i < 14; ++i) {
      asm volatile("global_load_async_to_lds_b128 %0, %1, off offset:%c2"
                   :
                   : "v"(ldsoff), "v"(ga), "i"(i * 16)
                   : "memory");
    }
  } else {
    uint4 z4 = {0u, 0u, 0u, 0u};
    uint4* dst = (uint4*)dstp;
#pragma unroll
    for (int i = 0; i < 14; ++i) dst[i] = z4;
  }
}

__device__ inline void wait_async_and_barrier() {
  asm volatile("s_wait_asynccnt 0x0" ::: "memory");
  __syncthreads();
}

// ---------------------------------------------------------------------------
// Pack hx = concat(h, x) to bf16 feature buffer; also pre-fill x-part of rx.
// ---------------------------------------------------------------------------
__global__ void pack_feat_kernel(const float* __restrict__ hf,
                                 const float* __restrict__ xf,
                                 unsigned short* __restrict__ featu,
                                 unsigned short* __restrict__ rxu,
                                 int Ntot) {
  size_t i = (size_t)blockIdx.x * blockDim.x + threadIdx.x;
  size_t total = (size_t)Ntot * CIN;
  if (i >= total) return;
  int n = (int)(i / CIN);
  int c = (int)(i - (size_t)n * CIN);
  float v = (c < 128) ? hf[(size_t)n * 128 + c] : xf[(size_t)n * 320 + (c - 128)];
  __bf16 b = (__bf16)v;
  unsigned short bits = __builtin_bit_cast(unsigned short, b);
  featu[i] = bits;
  if (c >= 128) rxu[i] = bits;
}

// ---------------------------------------------------------------------------
// Pack one weight tensor [27,448,128] f32 -> bf16 WMMA B-fragment layout.
// Block b = (k*14+kk)*8+ct, 32 lanes; lane writes its 8 contiguous dwords.
// B layout: lane col = ct*16 + l%16; VGPR j holds K = kk*32 + (j&3)*2 + (j>>2)*16 + (l>>4)*8, +1.
// ---------------------------------------------------------------------------
__global__ void pack_w_kernel(const float* __restrict__ W,
                              unsigned int* __restrict__ out) {
  int b = blockIdx.x;
  int lane = threadIdx.x;
  int ct = b & 7;
  int kkk = b >> 3;
  int kk = kkk % 14;
  int k  = kkk / 14;
  int col = ct * 16 + (lane & 15);
  int h = lane >> 4;
  unsigned int* dst = out + (size_t)b * 256 + (size_t)lane * 8;
#pragma unroll
  for (int j = 0; j < 8; ++j) {
    int kb = kk * 32 + (j & 3) * 2 + (j >> 2) * 16 + h * 8;
    float w0 = W[((size_t)k * CIN + kb) * COUT + col];
    float w1 = W[((size_t)k * CIN + kb + 1) * COUT + col];
    union { __bf16 bb[2]; unsigned int u; } pk;
    pk.bb[0] = (__bf16)w0;
    pk.bb[1] = (__bf16)w1;
    dst[j] = pk.u;
  }
}

// ---------------------------------------------------------------------------
// Kernel A: z/r convs fused (one gather, two WMMA accumulations per tile).
// Writes z (bf16 [N,128]) and r*h into rx[:, :128] (bf16 [N,448]).
// ---------------------------------------------------------------------------
__global__ __launch_bounds__(256) void conv_zr_kernel(
    const unsigned short* __restrict__ featu,
    const int* __restrict__ nbr_idx,
    const unsigned char* __restrict__ nbr_mask,
    const unsigned int* __restrict__ wz_pk,
    const unsigned int* __restrict__ wr_pk,
    const float* __restrict__ h_feat,
    unsigned short* __restrict__ z_u,
    unsigned short* __restrict__ rx_u,
    int Ntot) {
  __shared__ __align__(16) __bf16 As[ROWS_TILE * LDS_STRIDE];
  const __bf16* featbf = (const __bf16*)featu;
  __bf16* z_out = (__bf16*)z_u;
  __bf16* rx = (__bf16*)rx_u;

  int tile = blockIdx.x * ROWS_TILE;
  int tid = threadIdx.x;
  int wave = tid >> 5;
  int lane = tid & 31;

  v8f cz[4] = {};
  v8f cr[4] = {};

  for (int k = 0; k < 27; ++k) {
    __syncthreads();
    gather_tile_async(featbf, nbr_idx, nbr_mask, As, k, tile, tid, Ntot);
    wait_async_and_barrier();
    for (int kk = 0; kk < 14; ++kk) {
      v16bf bz = load_bfrag(wz_pk, k, kk, wave, lane);
      v16bf br = load_bfrag(wr_pk, k, kk, wave, lane);
#pragma unroll
      for (int rt = 0; rt < 4; ++rt) {
        v16bf a = load_afrag(As, rt * 16, kk, lane);
        cz[rt] = __builtin_amdgcn_wmma_f32_16x16x32_bf16(
            false, a, false, bz, (short)0, cz[rt], false, false);
        cr[rt] = __builtin_amdgcn_wmma_f32_16x16x32_bf16(
            false, a, false, br, (short)0, cr[rt], false, false);
      }
    }
  }

  // Epilogue: C layout -> lane col = l%16, VGPR j -> row j + 8*(l>>4).
  int colg = wave * 16 + (lane & 15);
  int h = lane >> 4;
#pragma unroll
  for (int rt = 0; rt < 4; ++rt) {
#pragma unroll
    for (int j = 0; j < 8; ++j) {
      int grow = tile + rt * 16 + h * 8 + j;
      if (grow < Ntot) {
        float zv = 1.f / (1.f + __expf(-cz[rt][j]));
        float rv = 1.f / (1.f + __expf(-cr[rt][j]));
        float hv = h_feat[(size_t)grow * COUT + colg];
        z_out[(size_t)grow * COUT + colg] = (__bf16)zv;
        rx[(size_t)grow * CIN + colg] = (__bf16)(rv * hv);
      }
    }
  }
}

// ---------------------------------------------------------------------------
// Kernel B: q conv over rx, fused tanh + GRU combine -> fp32 h_new.
// ---------------------------------------------------------------------------
__global__ __launch_bounds__(256) void conv_q_kernel(
    const unsigned short* __restrict__ rx_u,
    const int* __restrict__ nbr_idx,
    const unsigned char* __restrict__ nbr_mask,
    const unsigned int* __restrict__ wq_pk,
    const float* __restrict__ h_feat,
    const unsigned short* __restrict__ z_u,
    float* __restrict__ out,
    int Ntot) {
  __shared__ __align__(16) __bf16 As[ROWS_TILE * LDS_STRIDE];
  const __bf16* rxbf = (const __bf16*)rx_u;
  const __bf16* z_in = (const __bf16*)z_u;

  int tile = blockIdx.x * ROWS_TILE;
  int tid = threadIdx.x;
  int wave = tid >> 5;
  int lane = tid & 31;

  v8f cq[4] = {};

  for (int k = 0; k < 27; ++k) {
    __syncthreads();
    gather_tile_async(rxbf, nbr_idx, nbr_mask, As, k, tile, tid, Ntot);
    wait_async_and_barrier();
    for (int kk = 0; kk < 14; ++kk) {
      v16bf bq = load_bfrag(wq_pk, k, kk, wave, lane);
#pragma unroll
      for (int rt = 0; rt < 4; ++rt) {
        v16bf a = load_afrag(As, rt * 16, kk, lane);
        cq[rt] = __builtin_amdgcn_wmma_f32_16x16x32_bf16(
            false, a, false, bq, (short)0, cq[rt], false, false);
      }
    }
  }

  int colg = wave * 16 + (lane & 15);
  int h = lane >> 4;
#pragma unroll
  for (int rt = 0; rt < 4; ++rt) {
#pragma unroll
    for (int j = 0; j < 8; ++j) {
      int grow = tile + rt * 16 + h * 8 + j;
      if (grow < Ntot) {
        float qpre = cq[rt][j];
        float e = __expf(2.f * qpre);
        float q = (e - 1.f) / (e + 1.f);          // tanh
        float zv = (float)z_in[(size_t)grow * COUT + colg];
        float hv = h_feat[(size_t)grow * COUT + colg];
        out[(size_t)grow * COUT + colg] = (1.f - zv) * hv + zv * q;
      }
    }
  }
}

// ---------------------------------------------------------------------------
// Launch: pack feat + 3 weights, then conv_zr, then conv_q.
// ws layout (bytes, 256-aligned):
//   featbf  N*448*2
//   rxbf    N*448*2
//   zbf     N*128*2
//   wz/wr/wq packed  27*448*128*2 each   (total ~205 MiB at N=100k)
// ---------------------------------------------------------------------------
extern "C" void kernel_launch(void* const* d_in, const int* in_sizes, int n_in,
                              void* d_out, int out_size, void* d_ws, size_t ws_size,
                              hipStream_t stream) {
  const float* h_feat = (const float*)d_in[0];
  const float* x_feat = (const float*)d_in[1];
  const int* nbr_idx = (const int*)d_in[2];
  const unsigned char* nbr_mask = (const unsigned char*)d_in[3];
  const float* Wz = (const float*)d_in[4];
  const float* Wr = (const float*)d_in[5];
  const float* Wq = (const float*)d_in[6];

  int N = in_sizes[0] / COUT;

  char* ws = (char*)d_ws;
  size_t off = 0;
  auto take = [&](size_t bytes) {
    size_t o = off;
    off = (off + bytes + 255) & ~(size_t)255;
    return o;
  };
  size_t featB = (size_t)N * CIN * 2;
  size_t zB    = (size_t)N * COUT * 2;
  size_t wB    = (size_t)27 * CIN * COUT * 2;

  unsigned short* featbf = (unsigned short*)(ws + take(featB));
  unsigned short* rxbf   = (unsigned short*)(ws + take(featB));
  unsigned short* zbf    = (unsigned short*)(ws + take(zB));
  unsigned int*   wz_pk  = (unsigned int*)(ws + take(wB));
  unsigned int*   wr_pk  = (unsigned int*)(ws + take(wB));
  unsigned int*   wq_pk  = (unsigned int*)(ws + take(wB));

  // 1) pack features to bf16 (and x-part of rx)
  {
    size_t total = (size_t)N * CIN;
    int blocks = (int)((total + 255) / 256);
    pack_feat_kernel<<<blocks, 256, 0, stream>>>(h_feat, x_feat, featbf, rxbf, N);
  }
  // 2) pack weights into WMMA fragment layout
  {
    int blocks = 27 * 14 * 8;
    pack_w_kernel<<<blocks, 32, 0, stream>>>(Wz, wz_pk);
    pack_w_kernel<<<blocks, 32, 0, stream>>>(Wr, wr_pk);
    pack_w_kernel<<<blocks, 32, 0, stream>>>(Wq, wq_pk);
  }
  // 3) fused z/r convs
  {
    int blocks = (N + ROWS_TILE - 1) / ROWS_TILE;
    conv_zr_kernel<<<blocks, 256, 0, stream>>>(featbf, nbr_idx, nbr_mask,
                                               wz_pk, wr_pk, h_feat, zbf, rxbf, N);
  }
  // 4) q conv + GRU combine
  {
    int blocks = (N + ROWS_TILE - 1) / ROWS_TILE;
    conv_q_kernel<<<blocks, 256, 0, stream>>>(rxbf, nbr_idx, nbr_mask,
                                              wq_pk, h_feat, zbf,
                                              (float*)d_out, N);
  }
}
